// LSTMCell_44281112822474
// MI455X (gfx1250) — matrix operand order repaired and checked by reading, not compiled, not run on hardware
//
#include <hip/hip_runtime.h>

#define B_DIM 16384
#define IN_DIM 1024
#define H_DIM 1024
#define K_DIM 2048   // IN + H fused along K
#define KSTEPS (K_DIM / 32)

typedef __bf16 bf16_t;
typedef __attribute__((ext_vector_type(16))) __bf16 v16bf;
typedef __attribute__((ext_vector_type(8)))  __bf16 v8bf;
typedef __attribute__((ext_vector_type(8)))  float  v8f;
typedef __attribute__((ext_vector_type(4)))  float  v4f;

union Frag16 { v16bf v; v8bf h[2]; };

__device__ __forceinline__ float fsig(float x) {
    return __builtin_amdgcn_rcpf(1.0f + __expf(-x));
}
__device__ __forceinline__ float ftanh(float x) {
    return 2.0f * __builtin_amdgcn_rcpf(1.0f + __expf(-2.0f * x)) - 1.0f;
}

#define WMMA_BF16(A, Bf, C) \
    __builtin_amdgcn_wmma_f32_16x16x32_bf16(false, (A), false, (Bf), (short)0, (C), false, false)

// flat address of a __shared__ object -> LDS byte offset (aperture low bits are 0)
__device__ __forceinline__ unsigned int lds_off(const void* p) {
    return (unsigned int)(unsigned long long)p;
}
// one 16B async global->LDS transfer per lane (ASYNCcnt-tracked)
__device__ __forceinline__ void async_b128(unsigned int loff, const void* gptr) {
    asm volatile("global_load_async_to_lds_b128 %0, %1, off"
                 :: "v"(loff), "v"((unsigned long long)gptr)
                 : "memory");
}
__device__ __forceinline__ void wait_async0() {
    asm volatile("s_wait_asynccnt 0x0" ::: "memory");
}

// ---------------------------------------------------------------------------
// Prologue 1: pack X = [input | prev_h] to bf16, row-major [B][2048]
// ---------------------------------------------------------------------------
__global__ __launch_bounds__(256) void lstm_pack_x(
    const float* __restrict__ Xin, const float* __restrict__ Hin,
    bf16_t* __restrict__ Xb)
{
    const int b = blockIdx.x;
    const int t = threadIdx.x;
    const float* src = (t < 128) ? Xin : Hin;
    const int k = (t & 127) * 8;
    const float* p = src + (size_t)b * IN_DIM + k;
    v4f f0 = *(const v4f*)(p);
    v4f f1 = *(const v4f*)(p + 4);
    v8bf o;
#pragma unroll
    for (int e = 0; e < 4; ++e) {
        o[e]     = (bf16_t)f0[e];
        o[4 + e] = (bf16_t)f1[e];
    }
    *(v8bf*)&Xb[(size_t)b * K_DIM + (t >> 7) * IN_DIM + k] = o;
}

// ---------------------------------------------------------------------------
// Prologue 2: pack 8 fp32 weight matrices (K x N row-major) into bf16
// Wt[g][n][k] (N-major, K contiguous).  g order: 0=i, 1=f, 2=o, 3=u.
// ---------------------------------------------------------------------------
__global__ __launch_bounds__(256) void lstm_pack_w(
    const float* __restrict__ wxi, const float* __restrict__ whi,
    const float* __restrict__ wxf, const float* __restrict__ whf,
    const float* __restrict__ wxo, const float* __restrict__ who,
    const float* __restrict__ wxu, const float* __restrict__ whu,
    bf16_t* __restrict__ Wt)
{
    __shared__ float tile[32][33];
    const int g = blockIdx.z;
    const float* wx = (g == 0) ? wxi : (g == 1) ? wxf : (g == 2) ? wxo : wxu;
    const float* wh = (g == 0) ? whi : (g == 1) ? whf : (g == 2) ? who : whu;

    const int kglob0 = blockIdx.y * 32;
    const float* src = (kglob0 < IN_DIM) ? wx : wh;
    const int ks0 = kglob0 & (IN_DIM - 1);
    const int n0 = blockIdx.x * 32;

    const int tx = threadIdx.x & 31;
    const int ty = threadIdx.x >> 5;

#pragma unroll
    for (int p = 0; p < 4; ++p)
        tile[ty + 8 * p][tx] = src[(size_t)(ks0 + ty + 8 * p) * H_DIM + n0 + tx];

    __syncthreads();

#pragma unroll
    for (int p = 0; p < 4; ++p) {
        const int nl = ty + 8 * p;
        Wt[((size_t)g * H_DIM + n0 + nl) * K_DIM + kglob0 + tx] =
            (bf16_t)tile[tx][nl];
    }
}

// ---------------------------------------------------------------------------
// Main fused LSTM kernel.
// Block tile: M=128, N=64, 4 gates.  8 waves as 4(M) x 2(N);
// wave tile 32x32x4gates = 16 WMMAs/K-step.  LDS double buffered; staging is
// pure async global->LDS (6 x b128 per thread per K-step), one barrier/step.
// ---------------------------------------------------------------------------
__global__ __launch_bounds__(256) void lstm_wmma(
    const bf16_t* __restrict__ Xb, const bf16_t* __restrict__ Wt,
    const float* __restrict__ Cin,
    const float* __restrict__ b_i, const float* __restrict__ b_f,
    const float* __restrict__ b_o, const float* __restrict__ b_u,
    float* __restrict__ outH, float* __restrict__ outC)
{
    // stride 40 halfs (80 B): 16B-aligned vectors, conflict-free b128 access
    __shared__ __align__(16) bf16_t As[2][128][40];
    __shared__ __align__(16) bf16_t Bs[2][4][64][40];

    const int tid = threadIdx.x;
    const int m0 = blockIdx.x * 128;
    const int n0 = blockIdx.y * 64;

    const int lane = tid & 31;
    const int wave = tid >> 5;
    const int wm = wave & 3;        // 4 waves along M
    const int wn = wave >> 2;       // 2 waves along N

    const int arow0 = wm * 32 + (lane & 15);
    const int akb   = (lane >> 4) * 8;
    const int bcol0 = wn * 32 + (lane & 15);
    const int bkb   = (lane >> 4) * 8;

    // ---- async staging tasks (16B each) ----
    // A tile: 128 rows x 32 k = 512 transfers -> 2 per thread
    unsigned int offA[2][2];
    const bf16_t* gA[2];
#pragma unroll
    for (int q = 0; q < 2; ++q) {
        const int task = tid + q * 256;
        const int row = task >> 2;
        const int qu  = task & 3;
        gA[q] = Xb + (size_t)(m0 + row) * K_DIM + qu * 8;
        offA[0][q] = lds_off(&As[0][row][qu * 8]);
        offA[1][q] = lds_off(&As[1][row][qu * 8]);
    }
    // B tiles: 4 gates x 64 cols x 32 k = 1024 transfers -> 4 per thread
    unsigned int offB[2][4];
    const bf16_t* gB[4];
#pragma unroll
    for (int q = 0; q < 4; ++q) {
        const int task = tid + q * 256;
        const int g  = task >> 8;
        const int ix = task & 255;
        const int n  = ix >> 2;
        const int qu = ix & 3;
        gB[q] = Wt + ((size_t)g * H_DIM + n0 + n) * K_DIM + qu * 8;
        offB[0][q] = lds_off(&Bs[0][g][n][qu * 8]);
        offB[1][q] = lds_off(&Bs[1][g][n][qu * 8]);
    }

    v8f acc[4][2][2];
#pragma unroll
    for (int g = 0; g < 4; ++g)
#pragma unroll
        for (int ms = 0; ms < 2; ++ms)
#pragma unroll
            for (int ns = 0; ns < 2; ++ns)
#pragma unroll
                for (int j = 0; j < 8; ++j) acc[g][ms][ns][j] = 0.0f;

    auto issue_async = [&](int buf, int ktn) {
        const size_t koff = (size_t)ktn * 32;   // elements
#pragma unroll
        for (int q = 0; q < 2; ++q) async_b128(offA[buf][q], gA[q] + koff);
#pragma unroll
        for (int q = 0; q < 4; ++q) async_b128(offB[buf][q], gB[q] + koff);
    };

    auto compute = [&](int buf) {
        Frag16 a0, a1;
        a0.h[0] = *(const v8bf*)&As[buf][arow0][akb];
        a0.h[1] = *(const v8bf*)&As[buf][arow0][akb + 16];
        a1.h[0] = *(const v8bf*)&As[buf][arow0 + 16][akb];
        a1.h[1] = *(const v8bf*)&As[buf][arow0 + 16][akb + 16];
#pragma unroll
        for (int g = 0; g < 4; ++g) {
            Frag16 bf0, bf1;
            bf0.h[0] = *(const v8bf*)&Bs[buf][g][bcol0][bkb];
            bf0.h[1] = *(const v8bf*)&Bs[buf][g][bcol0][bkb + 16];
            bf1.h[0] = *(const v8bf*)&Bs[buf][g][bcol0 + 16][bkb];
            bf1.h[1] = *(const v8bf*)&Bs[buf][g][bcol0 + 16][bkb + 16];
            acc[g][0][0] = WMMA_BF16(a0.v, bf0.v, acc[g][0][0]);
            acc[g][0][1] = WMMA_BF16(a0.v, bf1.v, acc[g][0][1]);
            acc[g][1][0] = WMMA_BF16(a1.v, bf0.v, acc[g][1][0]);
            acc[g][1][1] = WMMA_BF16(a1.v, bf1.v, acc[g][1][1]);
        }
    };

    auto pipe_iter = [&](int buf, int kt) {
        wait_async0();        // my prefetch for `buf` has landed
        __syncthreads();      // everyone's landed; everyone done with buf^1
        if (kt + 1 < KSTEPS) issue_async(buf ^ 1, kt + 1);  // overlaps compute
        compute(buf);
    };

    // ---- pipeline ----
    issue_async(0, 0);
    for (int kt = 0; kt < KSTEPS; kt += 2) {
        pipe_iter(0, kt);
        pipe_iter(1, kt + 1);
    }

    // ---- fused epilogue: bias + nonlinearity + cell update ----
    // C/D layout: VGPR j -> M = j + 8*(lane>=16), N = lane%16
    const int rb = m0 + wm * 32 + ((lane >> 4) << 3);
#pragma unroll
    for (int ns = 0; ns < 2; ++ns) {
        const int c = n0 + wn * 32 + ns * 16 + (lane & 15);
        const float vbi = b_i[c], vbf = b_f[c], vbo = b_o[c], vbu = b_u[c];
#pragma unroll
        for (int ms = 0; ms < 2; ++ms) {
#pragma unroll
            for (int j = 0; j < 8; ++j) {
                const int r = rb + ms * 16 + j;
                const float gi = acc[0][ms][ns][j] + vbi;
                const float gf = acc[1][ms][ns][j] + vbf;
                const float go = acc[2][ms][ns][j] + vbo;
                const float gu = acc[3][ms][ns][j] + vbu;
                const float iv = fsig(gi);
                const float fv = fsig(gf);
                const float ov = fsig(go);
                const float uv = ftanh(gu);
                const float cp = Cin[(size_t)r * H_DIM + c];
                const float cn = fv * cp + iv * uv;
                const float hn = ov * ftanh(cn);
                outH[(size_t)r * H_DIM + c] = hn;
                outC[(size_t)r * H_DIM + c] = cn;
            }
        }
    }
}

// ---------------------------------------------------------------------------
extern "C" void kernel_launch(void* const* d_in, const int* in_sizes, int n_in,
                              void* d_out, int out_size, void* d_ws, size_t ws_size,
                              hipStream_t stream) {
    const float* input  = (const float*)d_in[0];
    const float* prev_h = (const float*)d_in[1];
    const float* prev_c = (const float*)d_in[2];
    const float* w_xi = (const float*)d_in[3];
    const float* w_hi = (const float*)d_in[4];
    const float* w_xf = (const float*)d_in[5];
    const float* w_hf = (const float*)d_in[6];
    const float* w_xu = (const float*)d_in[7];
    const float* w_hu = (const float*)d_in[8];
    const float* w_xo = (const float*)d_in[9];
    const float* w_ho = (const float*)d_in[10];
    const float* b_i  = (const float*)d_in[11];
    const float* b_f  = (const float*)d_in[12];
    const float* b_o  = (const float*)d_in[13];
    const float* b_u  = (const float*)d_in[14];

    bf16_t* Xb = (bf16_t*)d_ws;                            // 64 MB
    bf16_t* Wt = Xb + (size_t)B_DIM * K_DIM;               // 16 MB
    float* outH = (float*)d_out;
    float* outC = outH + (size_t)B_DIM * H_DIM;

    lstm_pack_x<<<B_DIM, 256, 0, stream>>>(input, prev_h, Xb);

    dim3 gP(H_DIM / 32, K_DIM / 32, 4);
    lstm_pack_w<<<gP, 256, 0, stream>>>(w_xi, w_hi, w_xf, w_hf,
                                        w_xo, w_ho, w_xu, w_hu, Wt);

    dim3 gG(B_DIM / 128, H_DIM / 64);
    lstm_wmma<<<gG, 256, 0, stream>>>(Xb, Wt, prev_c,
                                      b_i, b_f, b_o, b_u, outH, outC);
}
